// VQVAEGumbelMatrixLatent_40819369181463
// MI455X (gfx1250) — matrix operand order repaired and checked by reading, not compiled, not run on hardware
//
#include <hip/hip_runtime.h>
#include <hip/hip_bf16.h>

typedef __attribute__((ext_vector_type(16))) _Float16 v16h;
typedef __attribute__((ext_vector_type(8)))  _Float16 v8h;
typedef __attribute__((ext_vector_type(8)))  float    v8f;

#define NSV   20
#define NAV   1
#define FC0V  128
#define BSZ   4096
#define SNUM  16
#define INDIM 2688
#define E0D   1024
#define E1D   512
#define CDIM  64
#define KCODES 512
#define D0D   512
#define D1D   1024
#define ODIM  400

// ---------------------------------------------------------------------------
// Branch-free wave tile body: 16 x (JN*16) output, K stepped by 64.
// All loads of a K-chunk are issued together (single clause), then 2*JN WMMAs.
// ---------------------------------------------------------------------------
template<int JN>
__device__ __forceinline__
void gemm_wave_tile(const _Float16* __restrict__ A, const _Float16* __restrict__ BT,
                    float* __restrict__ C, const float* __restrict__ bias,
                    int N, int Kd, int rowBase, int colBase, int l15, int khalf) {
  v8f acc[JN] = {};

  const _Float16* aRow = A + (size_t)(rowBase + l15) * Kd + khalf * 16;
  const _Float16* bCol[JN];
  #pragma unroll
  for (int j = 0; j < JN; ++j)
    bCol[j] = BT + (size_t)(colBase + j * 16 + l15) * Kd + khalf * 16;

  for (int k0 = 0; k0 < Kd; k0 += 64) {
    // A fragments (contiguous K order, matching B): e0..15 = K[k+kh*16 .. +16)
    v8h a0lo = *(const v8h*)(aRow + k0);
    v8h a0hi = *(const v8h*)(aRow + k0 + 8);
    v8h a1lo = *(const v8h*)(aRow + k0 + 32);
    v8h a1hi = *(const v8h*)(aRow + k0 + 40);
    // B fragments for all JN subtiles: issued together, no control flow
    v16h b0[JN], b1[JN];
    #pragma unroll
    for (int j = 0; j < JN; ++j) {
      b0[j] = *(const v16h*)(bCol[j] + k0);
      b1[j] = *(const v16h*)(bCol[j] + k0 + 32);
    }
    // Stream-prefetch A ~1KB ahead (global_prefetch_b8; speculative => safe OOB)
    __builtin_prefetch(aRow + k0 + 512, 0, 1);
    v16h a0 = __builtin_shufflevector(a0lo, a0hi, 0,1,2,3,4,5,6,7,8,9,10,11,12,13,14,15);
    v16h a1 = __builtin_shufflevector(a1lo, a1hi, 0,1,2,3,4,5,6,7,8,9,10,11,12,13,14,15);
    // 2*JN back-to-back WMMAs, grouped by shared A fragment; the JN
    // accumulator chains are independent.
    #pragma unroll
    for (int j = 0; j < JN; ++j)
      acc[j] = __builtin_amdgcn_wmma_f32_16x16x32_f16(
          false, a0, false, b0[j], (short)0, acc[j], false, false);
    #pragma unroll
    for (int j = 0; j < JN; ++j)
      acc[j] = __builtin_amdgcn_wmma_f32_16x16x32_f16(
          false, a1, false, b1[j], (short)0, acc[j], false, false);
  }

  // C/D layout: VGPR r -> row = rowBase + khalf*8 + r, col = colBase + j*16 + l15
  #pragma unroll
  for (int j = 0; j < JN; ++j) {
    const int col = colBase + j * 16 + l15;
    const float bv = bias ? bias[col] : 0.0f;
    const int r0 = rowBase + khalf * 8;
    #pragma unroll
    for (int r = 0; r < 8; ++r)
      C[(size_t)(r0 + r) * N + col] = acc[j][r] + bv;
  }
}

// ---------------------------------------------------------------------------
// Generic WMMA GEMM: C(MxN, f32) = A(MxK, f16 row-major) * BT(NxK, f16)^T
// Block = 256 threads = 8 waves; block tile 64x128; wave tile 16x64.
// K multiple of 64; M multiple of 64; N multiple of 16. The N-edge dispatch
// is a scalar switch (readfirstlane) -> the hot path (JN=4) is branch-free
// and WMMA never runs under a partial EXEC mask.
// ---------------------------------------------------------------------------
__global__ __launch_bounds__(256)
void gemm_wmma_f16(const _Float16* __restrict__ A, const _Float16* __restrict__ BT,
                   float* __restrict__ C, const float* __restrict__ bias,
                   int M, int N, int Kd) {
  const int lane  = threadIdx.x & 31;
  const int wave  = threadIdx.x >> 5;
  const int waveM = wave & 3;
  const int waveN = wave >> 2;
  const int rowBase = blockIdx.y * 64 + waveM * 16;
  const int colBase = blockIdx.x * 128 + waveN * 64;
  const int l15   = lane & 15;
  const int khalf = lane >> 4;              // 0: K lo-half lanes, 1: K hi-half lanes

  int jv = (N - colBase) >> 4;
  jv = jv < 0 ? 0 : (jv > 4 ? 4 : jv);
  const int jmax = __builtin_amdgcn_readfirstlane(jv);

  switch (jmax) {
    case 4: gemm_wave_tile<4>(A, BT, C, bias, N, Kd, rowBase, colBase, l15, khalf); break;
    case 3: gemm_wave_tile<3>(A, BT, C, bias, N, Kd, rowBase, colBase, l15, khalf); break;
    case 2: gemm_wave_tile<2>(A, BT, C, bias, N, Kd, rowBase, colBase, l15, khalf); break;
    case 1: gemm_wave_tile<1>(A, BT, C, bias, N, Kd, rowBase, colBase, l15, khalf); break;
    default: break;                          // colBase >= N: nothing to do
  }
}

// ---------------------------------------------------------------------------
// Pack x = concat(feature, action) permuted to (BS, 21*128) and cast to f16
// ---------------------------------------------------------------------------
__global__ void pack_x(const float* __restrict__ feat, const float* __restrict__ act,
                       _Float16* __restrict__ X) {
  int idx = blockIdx.x * blockDim.x + threadIdx.x;
  if (idx >= BSZ * INDIM) return;
  int b = idx / INDIM;
  int r = idx % INDIM;
  int v = r / FC0V, c = r % FC0V;
  float val = (v < NSV) ? feat[((size_t)v * BSZ + b) * FC0V + c]
                        : act[(size_t)b * FC0V + c];
  X[idx] = (_Float16)val;
}

// W (Kd x N, f32 row-major) -> BT (N x Kd, f16 row-major)
__global__ void convert_transpose(const float* __restrict__ W, _Float16* __restrict__ BT,
                                  int Kd, int N) {
  int idx = blockIdx.x * blockDim.x + threadIdx.x;
  if (idx >= Kd * N) return;
  int k = idx / N, n = idx % N;
  BT[(size_t)n * Kd + k] = (_Float16)W[idx];
}

__global__ void convert_f16(const float* __restrict__ src, _Float16* __restrict__ dst, int n) {
  int i = blockIdx.x * blockDim.x + threadIdx.x;
  if (i < n) dst[i] = (_Float16)src[i];
}

// Per-column batch stats over 4096 rows: one thread per column (coalesced rows)
__global__ void bn_stats(const float* __restrict__ Y, float* __restrict__ mean,
                         float* __restrict__ rstd, int N) {
  int col = blockIdx.x * blockDim.x + threadIdx.x;
  if (col >= N) return;
  float s = 0.f, ss = 0.f;
  for (int r = 0; r < BSZ; ++r) {
    float v = Y[(size_t)r * N + col];
    s += v; ss += v * v;
  }
  float m = s * (1.0f / BSZ);
  float var = ss * (1.0f / BSZ) - m * m;    // biased (jnp.var default)
  mean[col] = m;
  rstd[col] = rsqrtf(var + 1e-5f);
}

__global__ void bn_apply_lrelu(const float* __restrict__ Y, const float* __restrict__ mean,
                               const float* __restrict__ rstd, const float* __restrict__ g,
                               const float* __restrict__ bb, _Float16* __restrict__ H, int N) {
  int idx = blockIdx.x * blockDim.x + threadIdx.x;
  if (idx >= BSZ * N) return;
  int col = idx % N;
  float v = (Y[idx] - mean[col]) * rstd[col] * g[col] + bb[col];
  v = (v >= 0.0f) ? v : 0.01f * v;
  H[idx] = (_Float16)v;
}

// ||codebook[:,k]||^2 for each of 512 codes (codebook is 64 x 512 row-major)
__global__ void codebook_norm(const float* __restrict__ cb, float* __restrict__ cbn) {
  int k = blockIdx.x * blockDim.x + threadIdx.x;
  if (k >= KCODES) return;
  float s = 0.f;
  for (int d = 0; d < CDIM; ++d) { float v = cb[(size_t)d * KCODES + k]; s += v * v; }
  cbn[k] = s;
}

// argmin_k (cbn[k] - 2*G[b,k]) then gather z_q = codebook[:, idx] as f16
__global__ __launch_bounds__(256)
void vq_select(const float* __restrict__ G, const float* __restrict__ cbn,
               const float* __restrict__ cb, _Float16* __restrict__ ZQ) {
  __shared__ float sd[256];
  __shared__ int   si[256];
  __shared__ int   sidx;
  int b = blockIdx.x;
  int t = threadIdx.x;
  float best = 3.0e38f; int bi = 0;
  for (int k = t; k < KCODES; k += 256) {
    float d = cbn[k] - 2.0f * G[(size_t)b * KCODES + k];
    if (d < best) { best = d; bi = k; }
  }
  sd[t] = best; si[t] = bi;
  __syncthreads();
  for (int s = 128; s > 0; s >>= 1) {
    if (t < s) {
      float od = sd[t + s]; int oi = si[t + s];
      if (od < sd[t] || (od == sd[t] && oi < si[t])) { sd[t] = od; si[t] = oi; }
    }
    __syncthreads();
  }
  if (t == 0) sidx = si[0];
  __syncthreads();
  if (t < CDIM) ZQ[(size_t)b * CDIM + t] = (_Float16)cb[(size_t)t * KCODES + sidx];
}

// Build la from Z (BS x 400), emit prob = sigmoid(la) and 16 hard Gumbel samples
__global__ void head_kernel(const float* __restrict__ Z, const float* __restrict__ U,
                            float* __restrict__ sample, float* __restrict__ prob) {
  const int TOT = BSZ * NSV * (NSV + NAV);
  int t = blockIdx.x * blockDim.x + threadIdx.x;
  if (t >= TOT) return;
  int b   = t / (NSV * (NSV + NAV));
  int rem = t % (NSV * (NSV + NAV));
  int i = rem / (NSV + NAV);
  int j = rem % (NSV + NAV);
  float la;
  if (j == i)      la = 100.0f;
  else if (j < i)  la = Z[(size_t)b * ODIM + i * NSV + j];        // tril(-1) part
  else             la = Z[(size_t)b * ODIM + i * NSV + (j - 1)];  // triu part (shift)
  prob[t] = 1.0f / (1.0f + expf(-la));
  #pragma unroll
  for (int s = 0; s < SNUM; ++s) {
    float u = U[(size_t)s * TOT + t];
    u = fminf(fmaxf(u, 1e-6f), 1.0f - 1e-6f);
    float logit = la + logf(u) - logf(1.0f - u);
    sample[(size_t)s * TOT + t] = (logit > 0.0f) ? 1.0f : 0.0f;   // hard forward value
  }
}

// ---------------------------------------------------------------------------
extern "C" void kernel_launch(void* const* d_in, const int* in_sizes, int n_in,
                              void* d_out, int out_size, void* d_ws, size_t ws_size,
                              hipStream_t stream) {
  (void)in_sizes; (void)n_in; (void)out_size; (void)ws_size;
  const float* feature  = (const float*)d_in[0];
  const float* action   = (const float*)d_in[1];
  const float* u_noise  = (const float*)d_in[2];
  const float* enc_W0   = (const float*)d_in[3];
  const float* enc_g0   = (const float*)d_in[4];
  const float* enc_b0   = (const float*)d_in[5];
  const float* enc_W1   = (const float*)d_in[6];
  const float* enc_g1   = (const float*)d_in[7];
  const float* enc_b1   = (const float*)d_in[8];
  const float* enc_Wf   = (const float*)d_in[9];
  const float* enc_bf   = (const float*)d_in[10];
  const float* codebook = (const float*)d_in[11];
  const float* dec_W0   = (const float*)d_in[12];
  const float* dec_g0   = (const float*)d_in[13];
  const float* dec_b0   = (const float*)d_in[14];
  const float* dec_W1   = (const float*)d_in[15];
  const float* dec_g1   = (const float*)d_in[16];
  const float* dec_b1   = (const float*)d_in[17];
  const float* dec_Wf   = (const float*)d_in[18];
  const float* dec_bf   = (const float*)d_in[19];

  char* ws = (char*)d_ws;
  size_t off = 0;
  auto take = [&](size_t bytes) -> void* {
    void* p = (void*)(ws + off);
    off = (off + bytes + 255) & ~(size_t)255;
    return p;
  };
  _Float16* X16   = (_Float16*)take((size_t)BSZ * INDIM * 2);
  _Float16* BT0   = (_Float16*)take((size_t)E0D * INDIM * 2);
  _Float16* BT1   = (_Float16*)take((size_t)E1D * E0D * 2);
  _Float16* BTf   = (_Float16*)take((size_t)CDIM * E1D * 2);
  _Float16* CBT   = (_Float16*)take((size_t)KCODES * CDIM * 2);
  _Float16* dBT0  = (_Float16*)take((size_t)D0D * CDIM * 2);
  _Float16* dBT1  = (_Float16*)take((size_t)D1D * D0D * 2);
  _Float16* dBTf  = (_Float16*)take((size_t)ODIM * D1D * 2);
  _Float16* H16   = (_Float16*)take((size_t)BSZ * 1024 * 2);
  _Float16* ZE16  = (_Float16*)take((size_t)BSZ * CDIM * 2);
  _Float16* ZQ16  = (_Float16*)take((size_t)BSZ * CDIM * 2);
  float*    Y     = (float*)take((size_t)BSZ * 1024 * 4);
  float*    meanB = (float*)take(1024 * 4);
  float*    rstdB = (float*)take(1024 * 4);
  float*    cbn   = (float*)take(KCODES * 4);

  auto cdiv = [](int a, int b) { return (a + b - 1) / b; };
  const dim3 blk(256);

  // Weight conversion / transposition (f32 -> f16, N x K layout)
  convert_transpose<<<cdiv(INDIM * E0D, 256), blk, 0, stream>>>(enc_W0, BT0, INDIM, E0D);
  convert_transpose<<<cdiv(E0D * E1D, 256), blk, 0, stream>>>(enc_W1, BT1, E0D, E1D);
  convert_transpose<<<cdiv(E1D * CDIM, 256), blk, 0, stream>>>(enc_Wf, BTf, E1D, CDIM);
  convert_transpose<<<cdiv(CDIM * KCODES, 256), blk, 0, stream>>>(codebook, CBT, CDIM, KCODES);
  convert_transpose<<<cdiv(CDIM * D0D, 256), blk, 0, stream>>>(dec_W0, dBT0, CDIM, D0D);
  convert_transpose<<<cdiv(D0D * D1D, 256), blk, 0, stream>>>(dec_W1, dBT1, D0D, D1D);
  convert_transpose<<<cdiv(D1D * ODIM, 256), blk, 0, stream>>>(dec_Wf, dBTf, D1D, ODIM);
  codebook_norm<<<2, blk, 0, stream>>>(codebook, cbn);
  pack_x<<<cdiv(BSZ * INDIM, 256), blk, 0, stream>>>(feature, action, X16);

  // Encoder
  gemm_wmma_f16<<<dim3(8, 64), blk, 0, stream>>>(X16, BT0, Y, (const float*)nullptr, BSZ, E0D, INDIM);
  bn_stats<<<E0D / 256, blk, 0, stream>>>(Y, meanB, rstdB, E0D);
  bn_apply_lrelu<<<cdiv(BSZ * E0D, 256), blk, 0, stream>>>(Y, meanB, rstdB, enc_g0, enc_b0, H16, E0D);
  gemm_wmma_f16<<<dim3(4, 64), blk, 0, stream>>>(H16, BT1, Y, (const float*)nullptr, BSZ, E1D, E0D);
  bn_stats<<<E1D / 256, blk, 0, stream>>>(Y, meanB, rstdB, E1D);
  bn_apply_lrelu<<<cdiv(BSZ * E1D, 256), blk, 0, stream>>>(Y, meanB, rstdB, enc_g1, enc_b1, H16, E1D);
  gemm_wmma_f16<<<dim3(1, 64), blk, 0, stream>>>(H16, BTf, Y, enc_bf, BSZ, CDIM, E1D);

  // VQ: G = z_e16 @ codebook, argmin(cbn - 2G), gather codebook column
  convert_f16<<<cdiv(BSZ * CDIM, 256), blk, 0, stream>>>(Y, ZE16, BSZ * CDIM);
  gemm_wmma_f16<<<dim3(4, 64), blk, 0, stream>>>(ZE16, CBT, Y, (const float*)nullptr, BSZ, KCODES, CDIM);
  vq_select<<<BSZ, blk, 0, stream>>>(Y, cbn, codebook, ZQ16);

  // Decoder
  gemm_wmma_f16<<<dim3(4, 64), blk, 0, stream>>>(ZQ16, dBT0, Y, (const float*)nullptr, BSZ, D0D, CDIM);
  bn_stats<<<D0D / 256, blk, 0, stream>>>(Y, meanB, rstdB, D0D);
  bn_apply_lrelu<<<cdiv(BSZ * D0D, 256), blk, 0, stream>>>(Y, meanB, rstdB, dec_g0, dec_b0, H16, D0D);
  gemm_wmma_f16<<<dim3(8, 64), blk, 0, stream>>>(H16, dBT1, Y, (const float*)nullptr, BSZ, D1D, D0D);
  bn_stats<<<D1D / 256, blk, 0, stream>>>(Y, meanB, rstdB, D1D);
  bn_apply_lrelu<<<cdiv(BSZ * D1D, 256), blk, 0, stream>>>(Y, meanB, rstdB, dec_g1, dec_b1, H16, D1D);
  gemm_wmma_f16<<<dim3(4, 64), blk, 0, stream>>>(H16, dBTf, Y, dec_bf, BSZ, ODIM, D1D);

  // Head: prob + 16 hard Gumbel-sigmoid samples; d_out = [sample | prob]
  const int TOT = BSZ * NSV * (NSV + NAV);
  float* out_sample = (float*)d_out;
  float* out_prob   = out_sample + (size_t)SNUM * TOT;
  head_kernel<<<cdiv(TOT, 256), blk, 0, stream>>>(Y, u_noise, out_sample, out_prob);
}
